// _PixelCNN_4793183502766
// MI455X (gfx1250) — matrix operand order repaired and checked by reading, not compiled, not run on hardware
//
#include <hip/hip_runtime.h>
#include <math.h>

#define NFC 160           // channels
#define HW  64            // H = W = 64
#define NB  32            // batch
#define NTAP3 5           // live taps in mask-B 3x3
#define CIP 168           // padded ci stride in LDS: 336B column stride (16B aligned, conflict-free)

typedef __attribute__((ext_vector_type(16))) __bf16 v16bf;
typedef __attribute__((ext_vector_type(8)))  float  v8f;

union Frag16 {
    v16bf v;
    unsigned int d[8];
    unsigned short u[16];
};

__device__ __forceinline__ unsigned short f2bf(float f) {
    unsigned int u = __float_as_uint(f);
    unsigned int r = u + 0x7FFFu + ((u >> 16) & 1u);   // round-to-nearest-even
    return (unsigned short)(r >> 16);
}

// async copy of 16B global -> LDS (per-lane addresses), tracked by ASYNCcnt
__device__ __forceinline__ void async_g2l_b128(unsigned int lds_byte_off,
                                               unsigned long long gaddr) {
    asm volatile("global_load_async_to_lds_b128 %0, %1, off"
                 :: "v"(lds_byte_off), "v"(gaddr) : "memory");
}
__device__ __forceinline__ void wait_async0() {
    asm volatile("s_wait_asynccnt 0x0" ::: "memory");
}

// ---------------------------------------------------------------------------
// K0: mask + convert residual weights to bf16 [layer][tap][co][ci], and
//     w_o1 to bf16 [co][ci].
// ---------------------------------------------------------------------------
__global__ __launch_bounds__(256)
void convert_weights_kernel(const float* __restrict__ wr0,
                            const float* __restrict__ wr1,
                            const float* __restrict__ wr2,
                            const float* __restrict__ wo1,
                            unsigned short* __restrict__ wrbf,
                            unsigned short* __restrict__ wo1bf)
{
    const int NW = 3 * NTAP3 * NFC * NFC;           // 384000
    int g = blockIdx.x * 256 + threadIdx.x;
    if (g < NW) {
        const int KY[NTAP3] = {0, 0, 0, 1, 1};
        const int KX[NTAP3] = {0, 1, 2, 0, 1};
        int layer = g / (NTAP3 * NFC * NFC);
        int r     = g % (NTAP3 * NFC * NFC);
        int tap   = r / (NFC * NFC);
        int r2    = r % (NFC * NFC);
        int co    = r2 / NFC;
        int ci    = r2 % NFC;
        const float* w = (layer == 0) ? wr0 : (layer == 1) ? wr1 : wr2;
        float v = w[((co * NFC + ci) * 3 + KY[tap]) * 3 + KX[tap]];
        wrbf[g] = f2bf(v);
    } else if (g < NW + NFC * NFC) {
        int r = g - NW;                              // [co*NFC + ci], OIHW 1x1
        wo1bf[r] = f2bf(wo1[r]);
    }
}

// ---------------------------------------------------------------------------
// K1: mask-A 7x7 input conv, C_in = 1 -> 160 channels (VALU, tiny FLOPs).
// Writes fp32 h (pre-relu, NCHW) and bf16 relu(h) in [b][y][x][ci] layout.
// ---------------------------------------------------------------------------
__global__ __launch_bounds__(256)
void inconv_kernel(const float* __restrict__ samples,
                   const float* __restrict__ w_in,
                   const float* __restrict__ b_in,
                   float* __restrict__ h0,
                   unsigned short* __restrict__ hbf0)
{
    int g = blockIdx.x * 256 + threadIdx.x;
    if (g >= NB * HW * HW * 20) return;
    int cog = g % 20;            // 20 groups of 8 output channels
    int p   = g / 20;
    int x   = p & 63;
    int y   = (p >> 6) & 63;
    int b   = p >> 12;
    int co0 = cog * 8;

    float acc[8];
#pragma unroll
    for (int j = 0; j < 8; ++j) acc[j] = b_in[co0 + j];

    for (int ky = 0; ky < 4; ++ky) {
        int yy = y + ky - 3;
        if (yy < 0) continue;
        int kxmax = (ky == 3) ? 3 : 7;
        for (int kx = 0; kx < kxmax; ++kx) {
            int xx = x + kx - 3;
            if (xx < 0 || xx > 63) continue;
            float s  = samples[(b << 12) + (yy << 6) + xx];
            float xv = 2.0f * s - 1.0f;
#pragma unroll
            for (int j = 0; j < 8; ++j)
                acc[j] = fmaf(w_in[(co0 + j) * 49 + ky * 7 + kx], xv, acc[j]);
        }
    }
    unsigned int pk[4];
#pragma unroll
    for (int j = 0; j < 8; ++j) {
        h0[((b * NFC + co0 + j) * HW + y) * HW + x] = acc[j];
        float r = acc[j] > 0.0f ? acc[j] : 0.0f;
        if (j & 1) pk[j >> 1] |= ((unsigned int)f2bf(r)) << 16;
        else       pk[j >> 1]  = (unsigned int)f2bf(r);
    }
    uint4 o; o.x = pk[0]; o.y = pk[1]; o.z = pk[2]; o.w = pk[3];
    *(uint4*)(hbf0 + ((size_t)((b * HW + y) * HW + x) * NFC + co0)) = o;
}

// ---------------------------------------------------------------------------
// K2: residual mask-B 3x3 block. fp32 h updated IN PLACE (conv taps read the
// bf16 relu'd activation tensor, so only same-position fp32 reads occur).
// relu(h_out) written as packed bf16 for the next layer.
// One block per (b, y): 320 threads = 10 waves; wave = M-tile (16 out ch),
// 4 N-tiles of 16 pixels. K = 5 taps x 160 ci, chunked by 32 (bf16 WMMA).
// Activation halo rows staged global->LDS with GLOBAL_LOAD_ASYNC_TO_LDS_B128.
// ---------------------------------------------------------------------------
__global__ __launch_bounds__(320)
void resblock_kernel(float* __restrict__ hio,                      // fp32 NCHW, in/out
                     const unsigned short* __restrict__ hbfin,     // bf16 [b][y][x][ci]
                     unsigned short* __restrict__ hbfout,          // bf16 [b][y][x][ci]
                     const unsigned short* __restrict__ wbf,       // [5][160][160] bf16
                     const float* __restrict__ bias)
{
    __shared__ __align__(16) unsigned short actL[2 * 66 * CIP];   // [row][col 0..65][ci]

    const int b   = blockIdx.x >> 6;
    const int y   = blockIdx.x & 63;
    const int tid = threadIdx.x;

    // zero halo columns (col 0 and col 65) for both rows
    for (int e = tid; e < 2 * 2 * CIP; e += 320) {
        int row  = e / (2 * CIP);
        int r    = e % (2 * CIP);
        int side = r / CIP;
        int ci   = r % CIP;
        actL[(row * 66 + (side ? 65 : 0)) * CIP + ci] = 0;
    }

    // stage relu'd bf16 rows y-1 (row 0) and y (row 1): 2*64 cols * 20 chunks of 16B
    const uint4 z4 = make_uint4(0u, 0u, 0u, 0u);
    if (y > 0) {
        unsigned long long src = (unsigned long long)(uintptr_t)hbfin +
            (unsigned long long)((size_t)((b * HW + (y - 1)) * HW) * NFC) * 2ull;
#pragma unroll
        for (int it = 0; it < 8; ++it) {               // 2560 chunks / 320 threads
            int e   = tid + it * 320;
            int row = e / 1280;
            int r   = e % 1280;
            int x   = r / 20;
            int c   = r % 20;
            unsigned int lds = (unsigned int)(uintptr_t)
                &actL[((row * 66 + x + 1) * CIP + c * 8)];
            unsigned long long ga = src +
                (unsigned long long)(((row * HW + x) * NFC + c * 8) * 2);
            async_g2l_b128(lds, ga);
        }
    } else {
        // row 0 (y-1 = -1): zeros; row 1: async copy of row y
        for (int e = tid; e < 1280; e += 320) {        // zero row 0
            int x = e / 20, c = e % 20;
            *(uint4*)&actL[((0 * 66 + x + 1) * CIP + c * 8)] = z4;
        }
        unsigned long long src = (unsigned long long)(uintptr_t)hbfin +
            (unsigned long long)((size_t)((b * HW + y) * HW) * NFC) * 2ull;
        for (int e = tid; e < 1280; e += 320) {
            int x = e / 20, c = e % 20;
            unsigned int lds = (unsigned int)(uintptr_t)
                &actL[((1 * 66 + x + 1) * CIP + c * 8)];
            unsigned long long ga = src +
                (unsigned long long)((x * NFC + c * 8) * 2);
            async_g2l_b128(lds, ga);
        }
    }
    wait_async0();
    __syncthreads();

    const int wave = tid >> 5;        // M-tile 0..9
    const int lane = tid & 31;
    const int lo   = lane & 15;
    const int hi   = lane >> 4;
    const int m    = wave * 16 + lo;  // output channel row of A

    v8f acc[4] = {};

    const int KY[NTAP3] = {0, 0, 0, 1, 1};
    const int KX[NTAP3] = {0, 1, 2, 0, 1};

    for (int tap = 0; tap < NTAP3; ++tap) {
        const int row = KY[tap];
        const int kx  = KX[tap];
        const unsigned int* wbase =
            (const unsigned int*)(wbf + (tap * NFC + m) * NFC);
        for (int c = 0; c < 5; ++c) {
            const int cb = 32 * c;
            // A fragment: 16x32 bf16, lane holds row m; pairs along K.
            Frag16 a;
            const int k0 = cb + 16 * hi;
#pragma unroll
            for (int i = 0; i < 4; ++i) a.d[i]     = wbase[(k0 + 2 * i) >> 1];
#pragma unroll
            for (int i = 0; i < 4; ++i) a.d[4 + i] = wbase[(k0 + 8 + 2 * i) >> 1];

#pragma unroll
            for (int nt = 0; nt < 4; ++nt) {
                const int x   = nt * 16 + lo;
                const int col = x + kx;              // padded column
                const unsigned int* bb = (const unsigned int*)
                    &actL[(row * 66 + col) * CIP + cb + 16 * hi];
                Frag16 bf;
#pragma unroll
                for (int j = 0; j < 8; ++j) bf.d[j] = bb[j];
                acc[nt] = __builtin_amdgcn_wmma_f32_16x16x32_bf16(
                    false, a.v, false, bf.v, (short)0, acc[nt], false, false);
            }
        }
    }

    // epilogue: residual + bias (fp32 in place) and packed relu bf16 for next layer
#pragma unroll
    for (int nt = 0; nt < 4; ++nt) {
        const int x = nt * 16 + lo;
        unsigned int pk[4];
#pragma unroll
        for (int v = 0; v < 8; ++v) {
            const int co  = wave * 16 + 8 * hi + v;
            const int idx = ((b * NFC + co) * HW + y) * HW + x;
            float val = hio[idx] + acc[nt][v] + bias[co];
            hio[idx] = val;
            float r = val > 0.0f ? val : 0.0f;
            if (v & 1) pk[v >> 1] |= ((unsigned int)f2bf(r)) << 16;
            else       pk[v >> 1]  = (unsigned int)f2bf(r);
        }
        uint4 o; o.x = pk[0]; o.y = pk[1]; o.z = pk[2]; o.w = pk[3];
        *(uint4*)(hbfout + ((size_t)((b * HW + y) * HW + x) * NFC
                            + wave * 16 + 8 * hi)) = o;
    }
}

// ---------------------------------------------------------------------------
// K3: head. out1 = relu(W_o1 * relu(h) + b_o1) via WMMA (K=160), then
// logit = w_o2 . out1 + b_o2, Bernoulli log-prob, reduce over one row.
// One block per (b, y). Deterministic LDS reduction (no float atomics).
// ---------------------------------------------------------------------------
__global__ __launch_bounds__(320)
void head_kernel(const unsigned short* __restrict__ hbf,   // relu'd bf16 [b][y][x][ci]
                 const float* __restrict__ samples,
                 const unsigned short* __restrict__ wo1bf, // [160][160] bf16
                 const float* __restrict__ b_o1,
                 const float* __restrict__ w_o2,           // [160]
                 const float* __restrict__ b_o2,           // [1]
                 float* __restrict__ partial)              // [NB][HW]
{
    __shared__ __align__(16) unsigned short actL[HW * CIP];  // [x][ci]
    __shared__ float wsumL[HW * 20];                          // [x][wave*2 + hi]
    __shared__ float lpL[HW];

    const int b   = blockIdx.x >> 6;
    const int y   = blockIdx.x & 63;
    const int tid = threadIdx.x;

    {   // async stage one row of relu'd activations (already bf16)
        unsigned long long src = (unsigned long long)(uintptr_t)hbf +
            (unsigned long long)((size_t)((b * HW + y) * HW) * NFC) * 2ull;
#pragma unroll
        for (int it = 0; it < 4; ++it) {                // 1280 chunks / 320 threads
            int e = tid + it * 320;
            int x = e / 20, c = e % 20;
            unsigned int lds = (unsigned int)(uintptr_t)&actL[x * CIP + c * 8];
            unsigned long long ga = src +
                (unsigned long long)((x * NFC + c * 8) * 2);
            async_g2l_b128(lds, ga);
        }
    }
    wait_async0();
    __syncthreads();

    const int wave = tid >> 5;
    const int lane = tid & 31;
    const int lo   = lane & 15;
    const int hi   = lane >> 4;
    const int m    = wave * 16 + lo;

    v8f acc[4] = {};
    const unsigned int* wbase = (const unsigned int*)(wo1bf + m * NFC);

    for (int c = 0; c < 5; ++c) {
        const int cb = 32 * c;
        Frag16 a;
        const int k0 = cb + 16 * hi;
#pragma unroll
        for (int i = 0; i < 4; ++i) a.d[i]     = wbase[(k0 + 2 * i) >> 1];
#pragma unroll
        for (int i = 0; i < 4; ++i) a.d[4 + i] = wbase[(k0 + 8 + 2 * i) >> 1];

#pragma unroll
        for (int nt = 0; nt < 4; ++nt) {
            const int x = nt * 16 + lo;
            const unsigned int* bb =
                (const unsigned int*)&actL[x * CIP + cb + 16 * hi];
            Frag16 bf;
#pragma unroll
            for (int j = 0; j < 8; ++j) bf.d[j] = bb[j];
            acc[nt] = __builtin_amdgcn_wmma_f32_16x16x32_bf16(
                false, a.v, false, bf.v, (short)0, acc[nt], false, false);
        }
    }

    // per-lane dot with w_o2 over its 8 output channels; scatter to LDS
#pragma unroll
    for (int nt = 0; nt < 4; ++nt) {
        const int x = nt * 16 + lo;
        float p = 0.0f;
#pragma unroll
        for (int v = 0; v < 8; ++v) {
            const int co = wave * 16 + 8 * hi + v;
            float o = acc[nt][v] + b_o1[co];
            o = o > 0.0f ? o : 0.0f;                 // relu after conv1x1
            p = fmaf(w_o2[co], o, p);
        }
        wsumL[x * 20 + wave * 2 + hi] = p;
    }
    __syncthreads();

    if (tid < HW) {
        float l = b_o2[0];
#pragma unroll
        for (int k = 0; k < 20; ++k) l += wsumL[tid * 20 + k];
        float s   = samples[(b << 12) + (y << 6) + tid];
        float t   = log1pf(__expf(-fabsf(l)));       // softplus(-|l|)
        float lp1 = (l < 0.0f ? l : 0.0f) - t;       // log sigmoid(l)
        float lp0 = (-l < 0.0f ? -l : 0.0f) - t;     // log sigmoid(-l)
        lpL[tid] = s * lp1 + (1.0f - s) * lp0;
    }
    __syncthreads();

    if (tid == 0) {
        float sum = 0.0f;
        for (int i = 0; i < HW; ++i) sum += lpL[i];
        partial[b * HW + y] = sum;
    }
}

// ---------------------------------------------------------------------------
// K4: deterministic final reduce over rows -> out[b]
// ---------------------------------------------------------------------------
__global__ __launch_bounds__(64)
void reduce_kernel(const float* __restrict__ partial, float* __restrict__ out)
{
    __shared__ float sm[64];
    const int b = blockIdx.x;
    const int t = threadIdx.x;
    sm[t] = partial[b * HW + t];
    __syncthreads();
    for (int s = 32; s > 0; s >>= 1) {
        if (t < s) sm[t] += sm[t + s];
        __syncthreads();
    }
    if (t == 0) out[b] = sm[0];
}

// ---------------------------------------------------------------------------
extern "C" void kernel_launch(void* const* d_in, const int* in_sizes, int n_in,
                              void* d_out, int out_size, void* d_ws, size_t ws_size,
                              hipStream_t stream)
{
    const float* samples = (const float*)d_in[0];
    const float* w_in    = (const float*)d_in[1];
    const float* b_in    = (const float*)d_in[2];
    const float* w_r0    = (const float*)d_in[3];
    const float* b_r0    = (const float*)d_in[4];
    const float* w_r1    = (const float*)d_in[5];
    const float* b_r1    = (const float*)d_in[6];
    const float* w_r2    = (const float*)d_in[7];
    const float* b_r2    = (const float*)d_in[8];
    const float* w_o1    = (const float*)d_in[9];
    const float* b_o1    = (const float*)d_in[10];
    const float* w_o2    = (const float*)d_in[11];
    const float* b_o2    = (const float*)d_in[12];
    float* out = (float*)d_out;

    const size_t HF  = (size_t)NB * NFC * HW * HW * sizeof(float);         // 83,886,080
    const size_t HBF = (size_t)NB * HW * HW * NFC * sizeof(unsigned short);// 41,943,040
    char* ws = (char*)d_ws;
    float* hio            = (float*)ws;                       // fp32 h (in-place residual)
    unsigned short* hbfA  = (unsigned short*)(ws + HF);       // relu'd bf16 activations
    unsigned short* hbfB  = (unsigned short*)(ws + HF + HBF);
    unsigned short* wrbf  = (unsigned short*)(ws + HF + 2 * HBF);
    unsigned short* wo1bf = wrbf + 3 * NTAP3 * NFC * NFC;
    float* partial        = (float*)(wo1bf + NFC * NFC);

    // K0: weight conversion
    {
        int total = 3 * NTAP3 * NFC * NFC + NFC * NFC;
        convert_weights_kernel<<<(total + 255) / 256, 256, 0, stream>>>(
            w_r0, w_r1, w_r2, w_o1, wrbf, wo1bf);
    }
    // K1: mask-A 7x7 input conv -> hio (fp32) + hbfA (relu bf16)
    {
        int total = NB * HW * HW * 20;
        inconv_kernel<<<(total + 255) / 256, 256, 0, stream>>>(
            samples, w_in, b_in, hio, hbfA);
    }
    // K2 x3: residual mask-B blocks; fp32 in place, bf16 relu ping-pong
    const int grid = NB * HW;  // one block per (b, y)
    resblock_kernel<<<grid, 320, 0, stream>>>(hio, hbfA, hbfB,
                                              wrbf + 0 * NTAP3 * NFC * NFC, b_r0);
    resblock_kernel<<<grid, 320, 0, stream>>>(hio, hbfB, hbfA,
                                              wrbf + 1 * NTAP3 * NFC * NFC, b_r1);
    resblock_kernel<<<grid, 320, 0, stream>>>(hio, hbfA, hbfB,
                                              wrbf + 2 * NTAP3 * NFC * NFC, b_r2);
    // K3: head (1x1 convs via WMMA + Bernoulli log-prob per row)
    head_kernel<<<grid, 320, 0, stream>>>(hbfB, samples, wo1bf, b_o1, w_o2, b_o2,
                                          partial);
    // K4: final reduce -> out[32]
    reduce_kernel<<<NB, 64, 0, stream>>>(partial, out);
}
